// NewSelfMultiheadAttention_39101382262913
// MI455X (gfx1250) — matrix operand order repaired and checked by reading, not compiled, not run on hardware
//
#include <hip/hip_runtime.h>
#include <hip/hip_bf16.h>
#include <math.h>

// ---------------------------------------------------------------------------
// Problem constants (from reference): B=4, T=S=1024, E=768, H=12, D=64, R=4
// SCALING = D^-0.5 = 0.125
// ---------------------------------------------------------------------------
#define BB   4
#define TT   1024
#define EE   768
#define HH   12
#define DD   64
#define RR   4

typedef _Float16 v8h  __attribute__((ext_vector_type(8)));
typedef _Float16 v16h __attribute__((ext_vector_type(16)));
typedef float    v4f  __attribute__((ext_vector_type(4)));
typedef float    v8f  __attribute__((ext_vector_type(8)));
typedef int      v4i  __attribute__((ext_vector_type(4)));

typedef __attribute__((address_space(1))) v4i* gv4i_ptr;  // global int4*
typedef __attribute__((address_space(3))) v4i* lv4i_ptr;  // LDS int4*

#if defined(__HIP_DEVICE_COMPILE__) && __has_builtin(__builtin_amdgcn_global_load_async_to_lds_b128)
#define HAVE_ASYNC_LDS 1
#else
#define HAVE_ASYNC_LDS 0
#endif

__device__ __forceinline__ v16h cat8(v8h lo, v8h hi) {
  return __builtin_shufflevector(lo, hi, 0,1,2,3,4,5,6,7,8,9,10,11,12,13,14,15);
}

// 16 consecutive f32 -> v16h (B-matrix layout: lane holds contiguous K run)
__device__ __forceinline__ v16h cvtB16(const float* __restrict__ p) {
  v16h r;
#pragma unroll
  for (int i = 0; i < 16; i += 4) {
    v4f t = *(const v4f*)(p + i);
    r[i+0]=(_Float16)t[0]; r[i+1]=(_Float16)t[1];
    r[i+2]=(_Float16)t[2]; r[i+3]=(_Float16)t[3];
  }
  return r;
}

// A-matrix layout from f32: elems 0..15 <- p[0..15]
__device__ __forceinline__ v16h cvtA16(const float* __restrict__ p) {
  v16h r;
#pragma unroll
  for (int i = 0; i < 8; i += 4) {
    v4f t0 = *(const v4f*)(p + i);
    v4f t1 = *(const v4f*)(p + 8 + i);
    r[i+0]=(_Float16)t0[0]; r[i+1]=(_Float16)t0[1];
    r[i+2]=(_Float16)t0[2]; r[i+3]=(_Float16)t0[3];
    r[8+i+0]=(_Float16)t1[0]; r[8+i+1]=(_Float16)t1[1];
    r[8+i+2]=(_Float16)t1[2]; r[8+i+3]=(_Float16)t1[3];
  }
  return r;
}

__device__ __forceinline__ v8f wmma_f16(v16h a, v16h b, v8f c) {
  return __builtin_amdgcn_wmma_f32_16x16x32_f16(false, a, false, b, (short)0, c,
                                                false, false);
}

// ---------------------------------------------------------------------------
// K1: qkv = query @ in_w^T + in_b ; split & repack to [B,H,T,D] f16.
// q additionally scaled by 0.125.
// Register-blocked: one wave computes a 16x64 output tile (A reused 4x).
// grid = (4096/16, 2304/64), block = 32
// ---------------------------------------------------------------------------
__global__ __launch_bounds__(32)
void qkv_proj_kernel(const float* __restrict__ query,
                     const float* __restrict__ in_w,
                     const float* __restrict__ in_b,
                     _Float16* __restrict__ qf,
                     _Float16* __restrict__ kf,
                     _Float16* __restrict__ vf) {
  const int lane = threadIdx.x;
  const int col  = lane & 15;
  const int half = lane >> 4;
  const int bt0  = blockIdx.x * 16;
  const int f0   = blockIdx.y * 64;     // 64-wide feature group (one head slab)
  const int akb  = half * 16;           // A-layout K base for this lane-half
  const int bkb  = half * 16;           // B-layout K base
  const int arow = bt0 + col;

  v8f acc[4] = {};
  for (int kb = 0; kb < EE; kb += 32) {
    v16h a = cvtA16(query + (size_t)arow * EE + kb + akb);
#pragma unroll
    for (int dt = 0; dt < 4; dt++) {
      v16h b = cvtB16(in_w + (size_t)(f0 + dt * 16 + col) * EE + kb + bkb);
      acc[dt] = wmma_f16(a, b, acc[dt]);
    }
  }

  const int which = f0 / EE;            // 0=q 1=k 2=v (64 | 768)
  const int fl    = f0 - which * EE;    // head slab base within q/k/v
  const int h     = fl >> 6;
  _Float16* dst = (which == 0) ? qf : (which == 1) ? kf : vf;
  const float scale = (which == 0) ? 0.125f : 1.0f;

#pragma unroll
  for (int dt = 0; dt < 4; dt++) {
    const float bias = in_b[f0 + dt * 16 + col];
    const int d = dt * 16 + col;
#pragma unroll
    for (int v = 0; v < 8; v++) {
      const int row = bt0 + v + half * 8;      // flat (b,t)
      const int b_  = row >> 10;
      const int t   = row & (TT - 1);
      const float val = (acc[dt][v] + bias) * scale;
      dst[(((size_t)(b_ * HH + h)) * TT + t) * DD + d] = (_Float16)val;
    }
  }
}

// ---------------------------------------------------------------------------
// K2: q_red[bh,t,:] = (8*qf[bh,t,:]) @ red_w^T + red_b     (qf carries x0.125)
// ---------------------------------------------------------------------------
__global__ __launch_bounds__(256)
void qred_kernel(const _Float16* __restrict__ qf,
                 const float* __restrict__ red_w,
                 const float* __restrict__ red_b,
                 float* __restrict__ q_red) {
  const int idx = blockIdx.x * 256 + threadIdx.x;   // bh*T + t
  if (idx >= BB * HH * TT) return;
  const _Float16* qp = qf + (size_t)idx * DD;
  float acc0 = red_b[0], acc1 = red_b[1], acc2 = red_b[2], acc3 = red_b[3];
#pragma unroll 8
  for (int dd = 0; dd < DD; dd++) {
    const float qv = 8.0f * (float)qp[dd];
    acc0 += qv * red_w[0 * DD + dd];
    acc1 += qv * red_w[1 * DD + dd];
    acc2 += qv * red_w[2 * DD + dd];
    acc3 += qv * red_w[3 * DD + dd];
  }
  float* o = q_red + (size_t)idx * RR;
  o[0] = acc0; o[1] = acc1; o[2] = acc2; o[3] = acc3;
}

// ---------------------------------------------------------------------------
// K3: flash attention with low-rank dynamic bias + attn_bias + key mask.
// One wave per (bh, 16-row query tile). grid = (T/16, B*H), block = 32.
// V chunks staged to LDS via GLOBAL_LOAD_ASYNC_TO_LDS_B128 when available
// (overlaps the QK^T WMMAs / softmax; synced with s_wait_asynccnt 0).
// ---------------------------------------------------------------------------
__global__ __launch_bounds__(32)
void attn_kernel(const _Float16* __restrict__ qf,
                 const _Float16* __restrict__ kf,
                 const _Float16* __restrict__ vf,
                 const float* __restrict__ q_red,
                 const float* __restrict__ kdyn,        // [B*H*T, R, T]
                 const float* __restrict__ abias,       // [B*H, T, T]
                 const unsigned char* __restrict__ kpm, // [B, T] bool
                 float* __restrict__ out_o) {           // [B, T, E]
  __shared__ _Float16 p_lds[16 * 32];   // P tile, row-major [t][s]
  __shared__ _Float16 v_lds[32 * 64];   // V chunk, row-major [s][d]

  const int lane = threadIdx.x;
  const int col  = lane & 15;
  const int half = lane >> 4;
  const int t0   = blockIdx.x * 16;
  const int bh   = blockIdx.y;
  const int b_   = bh / HH;
  const int h    = bh - b_ * HH;
  const int akb  = half * 16;
  const int bkb  = half * 16;

  const _Float16* qb = qf + ((size_t)bh * TT) * DD;
  const _Float16* kb = kf + ((size_t)bh * TT) * DD;
  const _Float16* vb = vf + ((size_t)bh * TT) * DD;

  // Q tile in A-layout registers (K = 64 -> two 32-chunks)
  const _Float16* qp = qb + (size_t)(t0 + col) * DD;
  const v16h aq0 = cat8(*(const v8h*)(qp + akb),      *(const v8h*)(qp + akb + 8));
  const v16h aq1 = cat8(*(const v8h*)(qp + akb + 32), *(const v8h*)(qp + akb + 40));

  // q_red for the 8 rows this lane-half owns
  float qr[8][RR];
#pragma unroll
  for (int v = 0; v < 8; v++) {
    const int t = t0 + v + half * 8;
    const float* p = q_red + ((size_t)bh * TT + t) * RR;
    qr[v][0] = p[0]; qr[v][1] = p[1]; qr[v][2] = p[2]; qr[v][3] = p[3];
  }

  float mrun[8], lrun[8];
  v8f oacc[4] = {};
#pragma unroll
  for (int v = 0; v < 8; v++) { mrun[v] = -1.0e30f; lrun[v] = 0.0f; }

  const float* abrow = abias + (size_t)bh * TT * TT;

  for (int sc = 0; sc < TT; sc += 32) {
    // ---- stage V chunk [32 s][64 d] into LDS ----
    {
      const _Float16* vp = vb + (size_t)(sc + lane) * DD;
#if HAVE_ASYNC_LDS
#pragma unroll
      for (int i = 0; i < DD; i += 8)
        __builtin_amdgcn_global_load_async_to_lds_b128(
            (gv4i_ptr)(vp + i),
            (lv4i_ptr)&v_lds[lane * DD + i],
            0, 0);
#else
#pragma unroll
      for (int i = 0; i < DD; i += 8)
        *(v8h*)(&v_lds[lane * DD + i]) = *(const v8h*)(vp + i);
#endif
    }

    // ---- scores for two 16-wide subtiles (overlaps the async V fetch) ----
    v8f csub[2];
#pragma unroll
    for (int sub = 0; sub < 2; sub++) {
      const int s0   = sc + sub * 16;
      const int scol = s0 + col;
      const _Float16* kp = kb + (size_t)scol * DD;
      const v16h bk0 = cat8(*(const v8h*)(kp + bkb),      *(const v8h*)(kp + bkb + 8));
      const v16h bk1 = cat8(*(const v8h*)(kp + 32 + bkb), *(const v8h*)(kp + 32 + bkb + 8));
      v8f c = {};
      c = wmma_f16(aq0, bk0, c);
      c = wmma_f16(aq1, bk1, c);

      const bool masked = kpm[b_ * TT + scol] != 0;
#pragma unroll
      for (int v = 0; v < 8; v++) {
        const int t = t0 + v + half * 8;
        const float* kd = kdyn + ((size_t)(bh * TT + t) * RR) * TT + scol;
        float dyn = qr[v][0] * __builtin_nontemporal_load(kd)
                  + qr[v][1] * __builtin_nontemporal_load(kd + TT)
                  + qr[v][2] * __builtin_nontemporal_load(kd + 2 * TT)
                  + qr[v][3] * __builtin_nontemporal_load(kd + 3 * TT);
        float bi = __builtin_nontemporal_load(abrow + (size_t)t * TT + scol);
        c[v] = masked ? -1.0e30f : (c[v] + dyn + bi);
      }
      csub[sub] = c;
    }

    // ---- online softmax (row stats via 16-lane butterfly in each half) ----
#pragma unroll
    for (int v = 0; v < 8; v++) {
      float tm = fmaxf(csub[0][v], csub[1][v]);
      tm = fmaxf(tm, __shfl_xor(tm, 1));
      tm = fmaxf(tm, __shfl_xor(tm, 2));
      tm = fmaxf(tm, __shfl_xor(tm, 4));
      tm = fmaxf(tm, __shfl_xor(tm, 8));
      const float mnew = fmaxf(mrun[v], tm);
      const float corr = __expf(mrun[v] - mnew);
      mrun[v] = mnew;
      oacc[0][v] *= corr; oacc[1][v] *= corr;
      oacc[2][v] *= corr; oacc[3][v] *= corr;

      const float p0 = __expf(csub[0][v] - mnew);
      const float p1 = __expf(csub[1][v] - mnew);
      float ts = p0 + p1;
      ts += __shfl_xor(ts, 1);
      ts += __shfl_xor(ts, 2);
      ts += __shfl_xor(ts, 4);
      ts += __shfl_xor(ts, 8);
      lrun[v] = lrun[v] * corr + ts;

      const int prow = v + half * 8;
      p_lds[prow * 32 + col]      = (_Float16)p0;
      p_lds[prow * 32 + 16 + col] = (_Float16)p1;
    }

#if HAVE_ASYNC_LDS
#if __has_builtin(__builtin_amdgcn_s_wait_asynccnt)
    __builtin_amdgcn_s_wait_asynccnt(0);
#else
    asm volatile("s_wait_asynccnt 0x0" ::: "memory");
#endif
#endif
    __syncthreads();   // P + V visible in LDS

    // ---- P (A-layout from LDS) x V (B-layout from LDS) ----
    const _Float16* pp = &p_lds[col * 32];
    const v16h ap = cat8(*(const v8h*)(pp + akb), *(const v8h*)(pp + akb + 8));
#pragma unroll
    for (int dt = 0; dt < 4; dt++) {
      v16h bv;
#pragma unroll
      for (int e = 0; e < 16; e++)
        bv[e] = v_lds[(bkb + e) * DD + dt * 16 + col];
      oacc[dt] = wmma_f16(ap, bv, oacc[dt]);
    }
    __syncthreads();   // ds reads drained before next iteration restages LDS
  }

  // ---- epilogue: normalize and scatter to [B, T, E] ----
#pragma unroll
  for (int v = 0; v < 8; v++) {
    const float rinv = 1.0f / lrun[v];
    const int t = t0 + v + half * 8;
    float* dst = out_o + ((size_t)b_ * TT + t) * EE + h * DD + col;
#pragma unroll
    for (int dt = 0; dt < 4; dt++)
      dst[dt * 16] = oacc[dt][v] * rinv;
  }
}

// ---------------------------------------------------------------------------
// K4: out = o @ out_w^T + out_b   (4096x768x768 GEMM, f32 in/out, f16 WMMA)
// Register-blocked 16x64 tile per wave.  grid = (4096/16, 768/64), block = 32
// ---------------------------------------------------------------------------
__global__ __launch_bounds__(32)
void out_proj_kernel(const float* __restrict__ o,
                     const float* __restrict__ out_w,
                     const float* __restrict__ out_b,
                     float* __restrict__ out) {
  const int lane = threadIdx.x;
  const int col  = lane & 15;
  const int half = lane >> 4;
  const int bt0  = blockIdx.x * 16;
  const int f0   = blockIdx.y * 64;
  const int akb  = half * 16;
  const int bkb  = half * 16;
  const int arow = bt0 + col;

  v8f acc[4] = {};
  for (int kbi = 0; kbi < EE; kbi += 32) {
    v16h a = cvtA16(o + (size_t)arow * EE + kbi + akb);
#pragma unroll
    for (int dt = 0; dt < 4; dt++) {
      v16h b = cvtB16(out_w + (size_t)(f0 + dt * 16 + col) * EE + kbi + bkb);
      acc[dt] = wmma_f16(a, b, acc[dt]);
    }
  }
#pragma unroll
  for (int dt = 0; dt < 4; dt++) {
    const int bcol = f0 + dt * 16 + col;
    const float bias = out_b[bcol];
#pragma unroll
    for (int v = 0; v < 8; v++) {
      const int row = bt0 + v + half * 8;
      out[(size_t)row * EE + bcol] = acc[dt][v] + bias;
    }
  }
}

// ---------------------------------------------------------------------------
// Host launcher
// ---------------------------------------------------------------------------
extern "C" void kernel_launch(void* const* d_in, const int* in_sizes, int n_in,
                              void* d_out, int out_size, void* d_ws, size_t ws_size,
                              hipStream_t stream) {
  const float* query  = (const float*)d_in[0];   // [B,T,E]
  const float* kdyn   = (const float*)d_in[1];   // [B*H*T, R, T]
  const unsigned char* kpm = (const unsigned char*)d_in[2]; // [B,T] bool
  const float* abias  = (const float*)d_in[3];   // [B*H, T, T]
  const float* in_w   = (const float*)d_in[4];   // [3E, E]
  const float* in_b   = (const float*)d_in[5];   // [3E]
  const float* red_w  = (const float*)d_in[6];   // [R, D]
  const float* red_b  = (const float*)d_in[7];   // [R]
  const float* out_w  = (const float*)d_in[8];   // [E, E]
  const float* out_b  = (const float*)d_in[9];   // [E]
  float* out = (float*)d_out;                    // [B,T,E]

  // Workspace layout (256B aligned regions)
  char* ws = (char*)d_ws;
  _Float16* qf    = (_Float16*)(ws);                           // 6,291,456 B
  _Float16* kf    = (_Float16*)(ws + 6291456);                 // 6,291,456 B
  _Float16* vf    = (_Float16*)(ws + 12582912);                // 6,291,456 B
  float*    q_red = (float*)   (ws + 18874368);                //   786,432 B
  float*    o_buf = (float*)   (ws + 19660800);                // 12,582,912 B
  (void)in_sizes; (void)n_in; (void)out_size; (void)ws_size;

  // K1: QKV projection (tiles: 4096/16 x 2304/64)
  qkv_proj_kernel<<<dim3(BB * TT / 16, 3 * EE / 64), 32, 0, stream>>>(
      query, in_w, in_b, qf, kf, vf);

  // K2: q_red
  qred_kernel<<<dim3((BB * HH * TT + 255) / 256), 256, 0, stream>>>(
      qf, red_w, red_b, q_red);

  // K3: attention (tiles: T/16 x B*H)
  attn_kernel<<<dim3(TT / 16, BB * HH), 32, 0, stream>>>(
      qf, kf, vf, q_red, kdyn, abias, kpm, o_buf);

  // K4: output projection
  out_proj_kernel<<<dim3(BB * TT / 16, EE / 64), 32, 0, stream>>>(
      o_buf, out_w, out_b, out);
}
